// Network_506806141128
// MI455X (gfx1250) — compile-verified, matrix operand order checked by reading, not run on hardware
//
#include <hip/hip_runtime.h>

// Problem constants (from reference)
#define BB 4
#define SS 96
#define NN 512
#define DD 8
#define HH 64
#define GG 256      // 4*H gate width
#define UU 96       // padded concat dim: 64 (h) + 8 (x) + 24 zero
#define UTP 520     // uT/xT row pitch in ushorts (bank-conflict pad, 16B aligned)
#define NWAVE 8
#define TARGET_NODE 0

typedef __attribute__((ext_vector_type(16))) __bf16          v16bf;
typedef __attribute__((ext_vector_type(8)))  float           v8f;
typedef __attribute__((ext_vector_type(16))) unsigned short  u16x16;
typedef __attribute__((ext_vector_type(8)))  unsigned short  u16x8;
typedef __attribute__((ext_vector_type(8)))  unsigned int    u32x8;
typedef __attribute__((ext_vector_type(4)))  unsigned int    u32x4;
typedef __attribute__((ext_vector_type(8)))  int             i32x8;
typedef __attribute__((ext_vector_type(4)))  int             i32x4;

struct __align__(16) Shared {
  unsigned short uT[HH][UTP];          // bf16 h^T  [j][n]           66,560 B
  unsigned short xT[16][UTP];          // bf16 x^T  [d][n], d>=8 zero 16,640 B
  unsigned short WT[GG][UU];           // bf16 [Wh;Wx]^T [gatecol][k] 49,152 B
  unsigned short mstg[NWAVE][16][UU];  // per-wave m staging          24,576 B
  float bias[GG];
  float hn0[HH];
  float t1[32];
};
static_assert(sizeof(Shared) <= 320 * 1024, "LDS budget");

// truncating f32 -> bf16 (1 VALU)
__device__ __forceinline__ unsigned short f2bf_t(float f) {
  return (unsigned short)(__float_as_uint(f) >> 16);
}
// pack two f32 -> packed bf16 pair with one v_perm_b32
__device__ __forceinline__ unsigned int bfpack2(float lo, float hi) {
  return __builtin_amdgcn_perm(__float_as_uint(hi), __float_as_uint(lo),
                               0x07060302u);
}
__device__ __forceinline__ float fsig(float z) {
  return 1.0f / (1.0f + __expf(-z));
}
__device__ __forceinline__ float ftanh(float z) {
  float e = __expf(-2.0f * z);
  return (1.0f - e) / (1.0f + e);
}

// ---- Tensor Data Mover: 1D contiguous f32 copy global -> LDS (6-arg builtin)
__device__ __forceinline__ void tdm_load_1d(const void* gsrc, unsigned int ldsOff,
                                            unsigned int nElems) {
  unsigned long long ga = (unsigned long long)gsrc;
  u32x4 g0;
  g0[0] = 1u;                                   // count=1, user desc
  g0[1] = ldsOff;                               // lds_addr
  g0[2] = (unsigned int)ga;                     // global_addr[31:0]
  g0[3] = (unsigned int)((ga >> 32) & 0x01FFFFFFu) | (2u << 30); // addr[56:32] | type=2
  i32x8 g1;
  g1[0] = (int)(2u << 16);                      // data_size = 4B
  g1[1] = (int)((nElems & 0xFFFFu) << 16);      // tensor_dim0 lo16
  g1[2] = (int)(((nElems >> 16) & 0xFFFFu) | (1u << 16)); // dim0 hi16 | tensor_dim1=1
  g1[3] = (int)((nElems & 0xFFFFu) << 16);      // tile_dim0 (<=65535)
  g1[4] = 0;                                    // tile_dim1/2 unused
  g1[5] = (int)nElems;                          // tensor_dim0_stride lo32
  g1[6] = 0;
  g1[7] = 0;
  i32x4 z4 = {0, 0, 0, 0};
  i32x8 z8 = {0, 0, 0, 0, 0, 0, 0, 0};
  __builtin_amdgcn_tensor_load_to_lds(g0, g1, z4, z4, z8, 0);
}

// ---- async global->LDS 16B copy (ASYNCcnt path)
__device__ __forceinline__ void async_g2l_b128(unsigned int ldsAddr,
                                               unsigned int byteOff,
                                               const void* sbase) {
  asm volatile("global_load_async_to_lds_b128 %0, %1, %2 offset:0"
               :: "v"(ldsAddr), "v"(byteOff), "s"(sbase) : "memory");
}
__device__ __forceinline__ void wait_async0() {
  asm volatile("s_wait_asynccnt 0" ::: "memory");
}

// B-fragment (and mstg A-fragment) from LDS row pointer, K layout per ISA 7.12.2
__device__ __forceinline__ v16bf ld_frag_lds(const unsigned short* rowp, int kbase) {
  u16x8 lo = *(const u16x8*)(rowp + kbase);
  u16x8 hi = *(const u16x8*)(rowp + kbase + 16);
  u16x16 c = __builtin_shufflevector(lo, hi, 0, 1, 2, 3, 4, 5, 6, 7,
                                     8, 9, 10, 11, 12, 13, 14, 15);
  return __builtin_bit_cast(v16bf, c);
}

// A-fragment from global f32 row (nontemporal stream); v_perm_b32 pairwise pack
__device__ __forceinline__ v16bf ld_afrag_global(const float* rowp) {
  u32x4 q0 = __builtin_nontemporal_load((const u32x4*)(rowp));
  u32x4 q1 = __builtin_nontemporal_load((const u32x4*)(rowp + 4));
  u32x4 q2 = __builtin_nontemporal_load((const u32x4*)(rowp + 16));
  u32x4 q3 = __builtin_nontemporal_load((const u32x4*)(rowp + 20));
  u32x8 r;
  r[0] = bfpack2(__uint_as_float(q0[0]), __uint_as_float(q0[1]));
  r[1] = bfpack2(__uint_as_float(q0[2]), __uint_as_float(q0[3]));
  r[2] = bfpack2(__uint_as_float(q1[0]), __uint_as_float(q1[1]));
  r[3] = bfpack2(__uint_as_float(q1[2]), __uint_as_float(q1[3]));
  r[4] = bfpack2(__uint_as_float(q2[0]), __uint_as_float(q2[1]));
  r[5] = bfpack2(__uint_as_float(q2[2]), __uint_as_float(q2[3]));
  r[6] = bfpack2(__uint_as_float(q3[0]), __uint_as_float(q3[1]));
  r[7] = bfpack2(__uint_as_float(q3[2]), __uint_as_float(q3[3]));
  return __builtin_bit_cast(v16bf, r);
}

__device__ __forceinline__ v8f wmma_bf16(v16bf a, v16bf b, v8f c) {
  return __builtin_amdgcn_wmma_f32_16x16x32_bf16(false, a, false, b,
                                                 (short)0, c, false, false);
}

__global__ void __launch_bounds__(256, 1)
graph_lstm_kernel(const float* __restrict__ x, const float* __restrict__ adj,
                  const float* __restrict__ h0, const float* __restrict__ c0,
                  const float* __restrict__ Wx, const float* __restrict__ Wh,
                  const float* __restrict__ bg, const float* __restrict__ W1,
                  const float* __restrict__ b1, const float* __restrict__ W2,
                  const float* __restrict__ b2, float* __restrict__ out,
                  float* __restrict__ cws, unsigned short* __restrict__ hws) {
  __shared__ Shared smem;

  const int b = blockIdx.x;
  const int tid = threadIdx.x;
  const int wave = tid >> 5;
  const int lane = tid & 31;
  const int lane16 = lane & 15;
  const int hf8 = (lane >= 16) ? 8 : 0;   // row-half offset in C/D layout
  const int koff = (lane >= 16) ? 8 : 0;  // K-half offset in A/B layout

  const float* adjB = adj + ((size_t)b * SS) * NN * NN;
  const float* xB   = x + ((size_t)b * SS) * NN * DD;
  const float* h0b  = h0 + (size_t)b * NN * HH;
  const float* c0b  = c0 + (size_t)b * NN * HH;
  float* cwsb = cws + (size_t)b * NN * HH;
  unsigned short* hwsb = hws + (size_t)b * NN * HH;
  float* xstage = (float*)&smem.mstg[0][0][0];  // 16 KB overlay on staging

  // ---------------- init: TDM weights into LDS staging (uT region) ------------
  float* wstage = (float*)&smem.uT[0][0];  // 83 KB contiguous region (uT+xT)
  if (wave == 0) {
    tdm_load_1d(Wh, (unsigned int)(unsigned long long)wstage, HH * GG);          // 64KB
    tdm_load_1d(Wx, (unsigned int)(unsigned long long)(wstage + HH * GG), DD * GG); // 8KB
  }
  __builtin_amdgcn_s_wait_tensorcnt(0);
  __syncthreads();

  // transpose+convert: WT[gatecol n][k] = (k<64 ? Wh[k][n] : k<72 ? Wx[k-64][n] : 0)
  for (int idx = tid; idx < GG * UU; idx += 256) {
    int n = idx / UU, k = idx % UU;
    float v = 0.0f;
    if (k < HH) v = wstage[k * GG + n];
    else if (k < HH + DD) v = wstage[HH * GG + (k - HH) * GG + n];
    smem.WT[n][k] = f2bf_t(v);
  }
  if (tid < GG) smem.bias[tid] = bg[tid];
  for (int idx = tid; idx < 8 * UTP; idx += 256)  // zero xT rows 8..15 (pad D->16)
    smem.xT[8 + idx / UTP][idx % UTP] = 0;
  __syncthreads();

  // ---------------- time loop --------------------------------------------------
  for (int t = 0; t < SS; ++t) {
    __syncthreads();  // staging/uT reuse fence from previous iteration

    // x_t slice (512x8 f32, contiguous 16KB) via TDM into xstage
    if (wave == 0)
      tdm_load_1d(xB + (size_t)t * NN * DD,
                  (unsigned int)(unsigned long long)xstage, NN * DD);

    // rebuild uT (bf16 h^T) for this step
    if (t == 0) {
      for (int idx = tid; idx < NN * HH; idx += 256) {
        int n = idx >> 6, j = idx & 63;
        smem.uT[j][n] = f2bf_t(h0b[idx]);
      }
    } else {
      // pure bf16 copy from scratch: async DMA global -> LDS, 16B chunks
#pragma unroll
      for (int i = 0; i < 16; ++i) {
        int c = tid + i * 256;           // 4096 chunks total
        int row = c >> 6;                // j
        int off8 = (c & 63) * 8;         // n start
        unsigned int lds =
            (unsigned int)(unsigned long long)&smem.uT[row][off8];
        async_g2l_b128(lds, (unsigned int)((row * NN + off8) * 2), hwsb);
      }
      wait_async0();
    }
    __builtin_amdgcn_s_wait_tensorcnt(0);
    __syncthreads();

    // xstage (f32 [n][d]) -> xT bf16 [d][n]
    for (int idx = tid; idx < NN * DD; idx += 256) {
      int n = idx >> 3, d = idx & 7;
      smem.xT[d][n] = f2bf_t(xstage[idx]);
    }
    __syncthreads();

    const float* adjt = adjB + (size_t)t * NN * NN;

    for (int rt = 0; rt < 4; ++rt) {
      const int n0 = (wave * 4 + rt) * 16;

      // -------- Phase A: m = a_t @ [h | x]  (5 j-tiles, K=512) --------
      v8f accA[5];
#pragma unroll
      for (int j = 0; j < 5; ++j) accA[j] = (v8f){0, 0, 0, 0, 0, 0, 0, 0};

      const float* rowbase = adjt + (size_t)(n0 + lane16) * NN + koff;
      for (int kc = 0; kc < 16; ++kc) {
        const int k0 = kc * 32;
        v16bf af = ld_afrag_global(rowbase + k0);
#pragma unroll
        for (int jt = 0; jt < 5; ++jt) {
          const unsigned short* rp =
              (jt < 4) ? &smem.uT[jt * 16 + lane16][0] : &smem.xT[lane16][0];
          v16bf bf = ld_frag_lds(rp, k0 + koff);
          accA[jt] = wmma_bf16(af, bf, accA[jt]);
        }
      }

      // stage m (bf16) for this wave's row-tile; zero pad cols 80..95
#pragma unroll
      for (int jt = 0; jt < 5; ++jt)
#pragma unroll
        for (int r = 0; r < 8; ++r)
          smem.mstg[wave][r + hf8][jt * 16 + lane16] = f2bf_t(accA[jt][r]);
#pragma unroll
      for (int r = 0; r < 8; ++r) smem.mstg[wave][r + hf8][80 + lane16] = 0;

      // -------- Phase B: g = m @ [Wh;Wx]^T + b ; gates; state update ----
      v16bf am[3];
#pragma unroll
      for (int kc = 0; kc < 3; ++kc)
        am[kc] = ld_frag_lds(&smem.mstg[wave][lane16][0], kc * 32 + koff);

#pragma unroll
      for (int jt = 0; jt < 4; ++jt) {
        v8f accB[4];
#pragma unroll
        for (int q = 0; q < 4; ++q) accB[q] = (v8f){0, 0, 0, 0, 0, 0, 0, 0};
#pragma unroll
        for (int kc = 0; kc < 3; ++kc) {
#pragma unroll
          for (int q = 0; q < 4; ++q) {
            const unsigned short* rp = &smem.WT[q * 64 + jt * 16 + lane16][0];
            v16bf bf = ld_frag_lds(rp, kc * 32 + koff);
            accB[q] = wmma_bf16(am[kc], bf, accB[q]);
          }
        }
        const int j = jt * 16 + lane16;
        const float bI = smem.bias[j], bF = smem.bias[64 + j];
        const float bO = smem.bias[128 + j], bG = smem.bias[192 + j];
        float hvals[8];
#pragma unroll
        for (int r = 0; r < 8; ++r) {
          const int n = n0 + r + hf8;
          float iv = fsig(accB[0][r] + bI);
          float fv = fsig(accB[1][r] + bF);
          float ov = fsig(accB[2][r] + bO);
          float gv = ftanh(accB[3][r] + bG);
          size_t ci = (size_t)n * HH + j;
          float cp = (t == 0) ? c0b[ci] : cwsb[ci];
          float cn = fv * cp + iv * gv;
          cwsb[ci] = cn;
          float hn = ov * ftanh(cn);
          hvals[r] = hn;
          if (n == TARGET_NODE) smem.hn0[j] = hn;
        }
        // packed bf16 h write-back: 8 row values per lane are contiguous in n
        unsigned int* hp =
            (unsigned int*)&hwsb[(size_t)j * NN + (n0 + hf8)];
#pragma unroll
        for (int rp2 = 0; rp2 < 4; ++rp2)
          hp[rp2] = bfpack2(hvals[2 * rp2], hvals[2 * rp2 + 1]);
      }
    }
    __threadfence();  // publish hws/cws for next step's async rebuild
  }

  // ---------------- readout MLP on node 0 -------------------------------------
  __syncthreads();
  if (tid < 32) {
    float a = b1[tid];
    for (int k = 0; k < HH; ++k) a += smem.hn0[k] * W1[k * 32 + tid];
    smem.t1[tid] = fmaxf(a, 0.0f);
  }
  __syncthreads();
  if (tid < 24) {
    float a = b2[tid];
    for (int j = 0; j < 32; ++j) a += smem.t1[j] * W2[j * 24 + tid];
    out[b * 24 + tid] = a;
  }
}

extern "C" void kernel_launch(void* const* d_in, const int* in_sizes, int n_in,
                              void* d_out, int out_size, void* d_ws, size_t ws_size,
                              hipStream_t stream) {
  (void)in_sizes; (void)n_in; (void)out_size; (void)ws_size;
  const float* x   = (const float*)d_in[0];
  const float* adj = (const float*)d_in[1];
  const float* h0  = (const float*)d_in[2];
  const float* c0  = (const float*)d_in[3];
  const float* Wx  = (const float*)d_in[4];
  const float* Wh  = (const float*)d_in[5];
  const float* bg  = (const float*)d_in[6];
  const float* W1  = (const float*)d_in[7];
  const float* b1  = (const float*)d_in[8];
  const float* W2  = (const float*)d_in[9];
  const float* b2  = (const float*)d_in[10];
  float* out = (float*)d_out;
  float* cws = (float*)d_ws;                                          // 512 KB f32 c-state
  unsigned short* hws =
      (unsigned short*)((char*)d_ws + (size_t)BB * NN * HH * 4);      // 256 KB bf16 h^T
  hipLaunchKernelGGL(graph_lstm_kernel, dim3(BB), dim3(256), 0, stream,
                     x, adj, h0, c0, Wx, Wh, bg, W1, b1, W2, b2, out, cws, hws);
}
